// DGMMC_spherical_54185307406409
// MI455X (gfx1250) — matrix operand order repaired and checked by reading, not compiled, not run on hardware
//
#include <hip/hip_runtime.h>

// Problem constants (match reference)
constexpr int BATCH = 8192;   // B
constexpr int DIM   = 1024;   // D
constexpr int CLS   = 128;    // C
constexpr int KMIX  = 8;      // K
constexpr int NCMP  = CLS * KMIX; // N = 1024
constexpr float LOG_2PI_F = 1.8378770664093453f;

// GEMM tiling
constexpr int BM = 128;
constexpr int BN = 128;
constexpr int BK = 64;
constexpr int LSTR = 72;  // padded LDS row stride in bf16 units (144 B, mult of 16)

typedef __attribute__((ext_vector_type(16))) __bf16 v16bf;
typedef __attribute__((ext_vector_type(8)))  float  v8f;
typedef __attribute__((ext_vector_type(4)))  int    i32x4;

union Frag {
    uint4 q[2];
    v16bf v;
};

#if __has_builtin(__builtin_amdgcn_global_load_async_to_lds_b128)
#define DGMM_USE_ASYNC 1
#endif

__device__ __forceinline__ void dgmm_wait_async0() {
#if __has_builtin(__builtin_amdgcn_s_wait_asynccnt)
    __builtin_amdgcn_s_wait_asynccnt(0);
#else
    asm volatile("s_wait_asynccnt 0x0" ::: "memory");
#endif
}

__device__ __forceinline__ void dgmm_cp16(const unsigned short* g, unsigned short* l) {
#ifdef DGMM_USE_ASYNC
    __builtin_amdgcn_global_load_async_to_lds_b128(
        (__attribute__((address_space(1))) i32x4*)g,
        (__attribute__((address_space(3))) i32x4*)l,
        0, 0);
#else
    *reinterpret_cast<uint4*>(l) = *reinterpret_cast<const uint4*>(g);
#endif
}

__device__ __forceinline__ unsigned short f2bf(float f) {
    unsigned int u = __float_as_uint(f);
    unsigned int r = u + 0x7fffu + ((u >> 16) & 1u);   // round-to-nearest-even
    return (unsigned short)(r >> 16);
}

// ---------------------------------------------------------------------------
// Kernel 0: f32 -> bf16 conversion (grid covers exactly count/4 elements)
// ---------------------------------------------------------------------------
__global__ void dgmm_cvt_kernel(const float* __restrict__ src,
                                unsigned short* __restrict__ dst) {
    size_t i = (size_t)blockIdx.x * 256 + threadIdx.x;
    float4 v = reinterpret_cast<const float4*>(src)[i];
    uint2 pk;
    pk.x = (unsigned int)f2bf(v.x) | ((unsigned int)f2bf(v.y) << 16);
    pk.y = (unsigned int)f2bf(v.z) | ((unsigned int)f2bf(v.w) << 16);
    reinterpret_cast<uint2*>(dst)[i] = pk;
}

// ---------------------------------------------------------------------------
// Kernel 1: per-component constants (exact f32)
//   coef[n] = -0.5*(D*log2pi + D*log(bw) + m_sq[n]/bw) + log_softmax_K(w)[n]
// Also computes log_softmax(priors) (block 0 only).
// ---------------------------------------------------------------------------
__global__ void dgmm_prep_kernel(const float* __restrict__ means,
                                 const float* __restrict__ bands,
                                 const float* __restrict__ weights,
                                 const float* __restrict__ priors,
                                 float* __restrict__ coef,
                                 float* __restrict__ inv_bw,
                                 float* __restrict__ logpri) {
    int n = blockIdx.x * 256 + threadIdx.x;   // grid covers exactly NCMP
    float bw = fminf(fmaxf(bands[n], 0.001f), 100.0f);
    float ibw = 1.0f / bw;

    float msq = 0.0f;
    const float* mp = means + (size_t)n * DIM;
    for (int d = 0; d < DIM; d += 4) {
        float4 v = *reinterpret_cast<const float4*>(mp + d);
        msq += v.x * v.x + v.y * v.y + v.z * v.z + v.w * v.w;
    }

    const float* wc = weights + ((n >> 3) << 3);
    float mx = wc[0];
    #pragma unroll
    for (int k = 1; k < KMIX; ++k) mx = fmaxf(mx, wc[k]);
    float s = 0.0f;
    #pragma unroll
    for (int k = 0; k < KMIX; ++k) s += expf(wc[k] - mx);
    float logw = weights[n] - (mx + logf(s));

    coef[n]   = -0.5f * ((float)DIM * LOG_2PI_F + (float)DIM * logf(bw) + msq * ibw) + logw;
    inv_bw[n] = ibw;

    if (blockIdx.x == 0 && threadIdx.x < CLS) {
        float pmx = priors[0];
        for (int c = 1; c < CLS; ++c) pmx = fmaxf(pmx, priors[c]);
        float ps = 0.0f;
        for (int c = 0; c < CLS; ++c) ps += expf(priors[c] - pmx);
        logpri[threadIdx.x] = priors[threadIdx.x] - (pmx + logf(ps));
    }
}

// ---------------------------------------------------------------------------
// Kernel 2: x row norms (one wave32 per row, exact f32)
// ---------------------------------------------------------------------------
__global__ void dgmm_xsq_kernel(const float* __restrict__ x,
                                float* __restrict__ xsq) {
    int lane = threadIdx.x & 31;
    int wid  = threadIdx.x >> 5;
    int b    = blockIdx.x * 8 + wid;
    const float4* row = reinterpret_cast<const float4*>(x + (size_t)b * DIM);
    float s = 0.0f;
    for (int i = lane; i < DIM / 4; i += 32) {
        float4 v = row[i];
        s += v.x * v.x + v.y * v.y + v.z * v.z + v.w * v.w;
    }
    #pragma unroll
    for (int off = 16; off > 0; off >>= 1) s += __shfl_xor(s, off, 32);
    if (lane == 0) xsq[b] = s;
}

// ---------------------------------------------------------------------------
// Kernel 3: bf16 WMMA GEMM (x @ means^T) with async global->LDS double
// buffering, fused log-prob + logsumexp over K=8 -> cls[B, C].
// Block tile 128x128, BK=64; 8 waves (4 M x 2 N), wave tile 32x64.
// ---------------------------------------------------------------------------
__global__ __launch_bounds__(256) void dgmm_gemm_kernel(
        const unsigned short* __restrict__ xb,
        const unsigned short* __restrict__ mb,
        const float* __restrict__ coef,
        const float* __restrict__ inv_bw,
        const float* __restrict__ xsq,
        float* __restrict__ cls) {
    __shared__ __align__(16) unsigned short As[2][BM * LSTR];
    __shared__ __align__(16) unsigned short Bs[2][BN * LSTR];

    const int t    = threadIdx.x;
    const int lane = t & 31;
    const int wid  = t >> 5;
    const int wm   = wid & 3;   // wave position along M (0..3)
    const int wn   = wid >> 2;  // wave position along N (0..1)
    const int gM0  = blockIdx.y * BM;
    const int gN0  = blockIdx.x * BN;

    v8f acc[2][4] = {};

    const int lm    = lane & 15;
    const int halfk = lane >> 4;

    // per-thread copy coordinates (16B chunks; 8 chunks per 64-wide row)
    int crow[4], ccol[4];
    #pragma unroll
    for (int i = 0; i < 4; ++i) {
        int u   = t + i * 256;          // 0..1023
        crow[i] = u >> 3;               // 0..127
        ccol[i] = (u & 7) * 8;          // 0..56 (bf16 units)
    }

    // prologue: copy chunk 0 into buffer 0
    #pragma unroll
    for (int i = 0; i < 4; ++i)
        dgmm_cp16(xb + (size_t)(gM0 + crow[i]) * DIM + ccol[i],
                  &As[0][crow[i] * LSTR + ccol[i]]);
    #pragma unroll
    for (int i = 0; i < 4; ++i)
        dgmm_cp16(mb + (size_t)(gN0 + crow[i]) * DIM + ccol[i],
                  &Bs[0][crow[i] * LSTR + ccol[i]]);
#ifdef DGMM_USE_ASYNC
    dgmm_wait_async0();
#endif
    __syncthreads();

    const int NCHUNK = DIM / BK;  // 16
    for (int chunk = 0; chunk < NCHUNK; ++chunk) {
        const int buf = chunk & 1;
        // kick off next chunk's copies into the other buffer
        if (chunk + 1 < NCHUNK) {
            const int k0 = (chunk + 1) * BK;
            #pragma unroll
            for (int i = 0; i < 4; ++i)
                dgmm_cp16(xb + (size_t)(gM0 + crow[i]) * DIM + k0 + ccol[i],
                          &As[buf ^ 1][crow[i] * LSTR + ccol[i]]);
            #pragma unroll
            for (int i = 0; i < 4; ++i)
                dgmm_cp16(mb + (size_t)(gN0 + crow[i]) * DIM + k0 + ccol[i],
                          &Bs[buf ^ 1][crow[i] * LSTR + ccol[i]]);
        }

        // compute on current buffer: 2 k-steps of K=32, 2x4 tiles each
        #pragma unroll
        for (int ks = 0; ks < 2; ++ks) {
            const int kofs = ks * 32;
            Frag a[2], fb[4];
            #pragma unroll
            for (int i = 0; i < 2; ++i) {
                const unsigned short* p =
                    &As[buf][(wm * 32 + i * 16 + lm) * LSTR + kofs + halfk * 8];
                a[i].q[0] = *reinterpret_cast<const uint4*>(p);       // K = 8h+0..7
                a[i].q[1] = *reinterpret_cast<const uint4*>(p + 16);  // K = 16+8h+0..7
            }
            #pragma unroll
            for (int j = 0; j < 4; ++j) {
                const unsigned short* p =
                    &Bs[buf][(wn * 64 + j * 16 + lm) * LSTR + kofs + halfk * 8];
                fb[j].q[0] = *reinterpret_cast<const uint4*>(p);      // K = 8h+0..7
                fb[j].q[1] = *reinterpret_cast<const uint4*>(p + 16); // K = 16+8h+0..7
            }
            #pragma unroll
            for (int i = 0; i < 2; ++i)
                #pragma unroll
                for (int j = 0; j < 4; ++j)
                    acc[i][j] = __builtin_amdgcn_wmma_f32_16x16x32_bf16(
                        false, a[i].v, false, fb[j].v,
                        (short)0, acc[i][j], false, false);
        }

#ifdef DGMM_USE_ASYNC
        dgmm_wait_async0();
#endif
        __syncthreads();
    }

    // --- Epilogue: log-prob + logsumexp over K=8 lanes -> cls[b, c] ---
    const int col   = lane & 15;
    const int halfm = lane >> 4;
    #pragma unroll
    for (int i = 0; i < 2; ++i) {
        #pragma unroll
        for (int j = 0; j < 4; ++j) {
            int ng = gN0 + wn * 64 + j * 16 + col;     // global component idx
            float cf  = coef[ng];
            float ib  = inv_bw[ng];
            int bbase = gM0 + wm * 32 + i * 16 + halfm * 8;
            #pragma unroll
            for (int r = 0; r < 8; ++r) {
                int b = bbase + r;
                float val = cf + ib * acc[i][j][r] - 0.5f * ib * xsq[b];
                float mx = val;
                mx = fmaxf(mx, __shfl_xor(mx, 1, 32));
                mx = fmaxf(mx, __shfl_xor(mx, 2, 32));
                mx = fmaxf(mx, __shfl_xor(mx, 4, 32));
                float se = __expf(val - mx);
                se += __shfl_xor(se, 1, 32);
                se += __shfl_xor(se, 2, 32);
                se += __shfl_xor(se, 4, 32);
                float lse = mx + __logf(se);
                if ((lane & 7) == 0)
                    cls[(size_t)b * CLS + (ng >> 3)] = lse;
            }
        }
    }
}

// ---------------------------------------------------------------------------
// Kernel 4: add log-priors, normalize over classes (logsumexp over C=128)
// ---------------------------------------------------------------------------
__global__ void dgmm_finalize_kernel(const float* __restrict__ cls,
                                     const float* __restrict__ logpri,
                                     float* __restrict__ out) {
    __shared__ float sm[CLS];
    int c = threadIdx.x;
    size_t b = blockIdx.x;
    float v = cls[b * CLS + c] + logpri[c];
    sm[c] = v;
    __syncthreads();
    for (int s = CLS / 2; s > 0; s >>= 1) {
        if (c < s) sm[c] = fmaxf(sm[c], sm[c + s]);
        __syncthreads();
    }
    float mx = sm[0];
    __syncthreads();
    sm[c] = __expf(v - mx);
    __syncthreads();
    for (int s = CLS / 2; s > 0; s >>= 1) {
        if (c < s) sm[c] += sm[c + s];
        __syncthreads();
    }
    float lse = mx + __logf(sm[0]);
    out[b * CLS + c] = v - lse;
}

// ---------------------------------------------------------------------------
extern "C" void kernel_launch(void* const* d_in, const int* in_sizes, int n_in,
                              void* d_out, int out_size, void* d_ws, size_t ws_size,
                              hipStream_t stream) {
    (void)in_sizes; (void)n_in; (void)out_size; (void)ws_size;
    const float* x       = (const float*)d_in[0];
    const float* means   = (const float*)d_in[1];
    const float* bands   = (const float*)d_in[2];
    const float* weights = (const float*)d_in[3];
    const float* priors  = (const float*)d_in[4];
    float* out = (float*)d_out;

    float* ws     = (float*)d_ws;
    float* cls    = ws;                                   // B*C
    float* xsq    = cls + (size_t)BATCH * CLS;            // B
    float* coef   = xsq + BATCH;                          // N
    float* inv_bw = coef + NCMP;                          // N
    float* logpri = inv_bw + NCMP;                        // C
    unsigned short* xb = (unsigned short*)(logpri + CLS); // B*D bf16
    unsigned short* mbf = xb + (size_t)BATCH * DIM;       // N*D bf16

    dgmm_cvt_kernel<<<(BATCH * (DIM / 4)) / 256, 256, 0, stream>>>(x, xb);
    dgmm_cvt_kernel<<<(NCMP * (DIM / 4)) / 256, 256, 0, stream>>>(means, mbf);

    dgmm_prep_kernel<<<NCMP / 256, 256, 0, stream>>>(
        means, bands, weights, priors, coef, inv_bw, logpri);

    dgmm_xsq_kernel<<<BATCH / 8, 256, 0, stream>>>(x, xsq);

    dim3 grid(NCMP / BN, BATCH / BM);   // (8, 64)
    dgmm_gemm_kernel<<<grid, 256, 0, stream>>>(
        xb, mbf, coef, inv_bw, xsq, cls);

    dgmm_finalize_kernel<<<BATCH, CLS, 0, stream>>>(cls, logpri, out);
}
